// TemporalAttention_86380382257197
// MI455X (gfx1250) — matrix operand order repaired and checked by reading, compiled, not run on hardware
//
#include <hip/hip_runtime.h>
#include <hip/hip_bf16.h>

typedef __attribute__((ext_vector_type(16))) __bf16 v16bf;
typedef __attribute__((ext_vector_type(8)))  __bf16 v8bf;
typedef __attribute__((ext_vector_type(8)))  float  v8f;

#define NB  2
#define NT  2048
#define NDM 2048
#define NH  16
#define NHD 128

// Packed-fragment LDS layout: each 16x16x32 operand tile is stored as
// 32 lanes x 16 bf16 (32 B per lane), so a fragment load is one contiguous
// 32-byte read per lane (2x ds_load_b128) and WMMA consumes it directly.
//   A-tile lane = 16*half + rowIdx, element j <-> k = j + 16*half
//     inverse: half=k>>4, j=k&15                 (16-k rows contiguous)
//   B-tile lane = 16*half + colIdx, element j <-> k = j + 16*half
//     inverse: half=k>>4, j=k&15                 (16-k columns contiguous)

// ---------------------------------------------------------------------------
// 128x128-tile bf16 GEMM, fp32 accumulate.  MODE 0: C row-major [M x N].
// MODE 1: scatter C into [B, H, T, HD] head-major layout (fused transpose).
// ---------------------------------------------------------------------------
template <int MODE>
__global__ __launch_bounds__(256) void gemm128_bf16(
    const float* __restrict__ A, const float* __restrict__ Bm,
    float* __restrict__ C, int M, int N, int K) {
  __shared__ __bf16 Apack[8 * 32 * 16];   // 8 m-tiles, packed A fragments
  __shared__ __bf16 Bpack[8 * 32 * 16];   // 8 n-tiles, packed B fragments

  const int tid  = threadIdx.x;
  const int lane = tid & 31;
  const int wave = tid >> 5;
  const int wm   = wave >> 1;   // 0..3 : 32 rows each
  const int wn   = wave & 1;    // 0..1 : 64 cols each
  const int mBlk = blockIdx.y * 128;
  const int nBlk = blockIdx.x * 128;

  const v8f zero = {};
  v8f acc[2][4];
#pragma unroll
  for (int i = 0; i < 2; ++i)
#pragma unroll
    for (int n = 0; n < 4; ++n) acc[i][n] = zero;

  const int arow  = tid >> 1;           // A: one row, 16 consecutive k
  const int aseg  = (tid & 1) * 16;
  const int bn    = tid & 127;          // B: one column, 16 consecutive k
  const int khalf = tid >> 7;

  for (int kt = 0; kt < K; kt += 32) {
    {  // stage A (row loads, packed stores)
      const float* ap = A + (size_t)(mBlk + arow) * K + kt + aseg;
      float4 v0 = *reinterpret_cast<const float4*>(ap + 0);
      float4 v1 = *reinterpret_cast<const float4*>(ap + 4);
      float4 v2 = *reinterpret_cast<const float4*>(ap + 8);
      float4 v3 = *reinterpret_cast<const float4*>(ap + 12);
      v8bf p0, p1;
      p0[0] = (__bf16)v0.x; p0[1] = (__bf16)v0.y; p0[2] = (__bf16)v0.z; p0[3] = (__bf16)v0.w;
      p0[4] = (__bf16)v1.x; p0[5] = (__bf16)v1.y; p0[6] = (__bf16)v1.z; p0[7] = (__bf16)v1.w;
      p1[0] = (__bf16)v2.x; p1[1] = (__bf16)v2.y; p1[2] = (__bf16)v2.z; p1[3] = (__bf16)v2.w;
      p1[4] = (__bf16)v3.x; p1[5] = (__bf16)v3.y; p1[6] = (__bf16)v3.z; p1[7] = (__bf16)v3.w;
      const int mt = arow >> 4;
      *reinterpret_cast<v8bf*>(Apack + (mt * 32 + aseg + (arow & 15)) * 16 + 0) = p0;
      *reinterpret_cast<v8bf*>(Apack + (mt * 32 + aseg + (arow & 15)) * 16 + 8) = p1;
      if (kt + 32 < K) __builtin_prefetch(ap + 32, 0, 0);  // global_prefetch_b8
    }
    {  // stage B (column loads -> contiguous packed stores)
      const float* bp = Bm + (size_t)(kt + khalf * 16) * N + nBlk + bn;
      float cv[16];
#pragma unroll
      for (int u = 0; u < 16; ++u) cv[u] = bp[(size_t)u * N];
      v8bf p0, p1;
#pragma unroll
      for (int u = 0; u < 8; ++u) { p0[u] = (__bf16)cv[u]; p1[u] = (__bf16)cv[u + 8]; }
      __bf16* dst = Bpack + ((bn >> 4) * 32 + 16 * khalf + (bn & 15)) * 16;
      *reinterpret_cast<v8bf*>(dst)     = p0;
      *reinterpret_cast<v8bf*>(dst + 8) = p1;
      if (kt + 32 < K) __builtin_prefetch(bp + (size_t)32 * N, 0, 0);
    }
    __syncthreads();

    v16bf af[2];
#pragma unroll
    for (int i = 0; i < 2; ++i)
      af[i] = *reinterpret_cast<const v16bf*>(Apack + ((wm * 2 + i) * 32 + lane) * 16);
    v16bf bf[4];
#pragma unroll
    for (int n = 0; n < 4; ++n)
      bf[n] = *reinterpret_cast<const v16bf*>(Bpack + ((wn * 4 + n) * 32 + lane) * 16);
#pragma unroll
    for (int i = 0; i < 2; ++i)
#pragma unroll
      for (int n = 0; n < 4; ++n)
        acc[i][n] = __builtin_amdgcn_wmma_f32_16x16x32_bf16(
            false, af[i], false, bf[n], (short)0, acc[i][n], false, false);
    __syncthreads();
  }

  const int half = lane >> 4;
  const int idx  = lane & 15;
#pragma unroll
  for (int i = 0; i < 2; ++i) {
    const int row0 = mBlk + wm * 32 + i * 16 + 8 * half;
#pragma unroll
    for (int n = 0; n < 4; ++n) {
      const int col = nBlk + wn * 64 + n * 16 + idx;
#pragma unroll
      for (int r = 0; r < 8; ++r) {
        const int row = row0 + r;
        const float val = acc[i][n][r];
        if (MODE == 0) {
          C[(size_t)row * N + col] = val;
        } else {
          const int b  = row >> 11;
          const int t  = row & (NT - 1);
          const int h  = col >> 7;
          const int hd = col & (NHD - 1);
          C[(((size_t)(b * NH + h)) * NT + t) * NHD + hd] = val;
        }
      }
    }
  }
}

// ---------------------------------------------------------------------------
// RoPE in place on q (workspace) and k (d_out), both [B,H,T,HD]
// ---------------------------------------------------------------------------
__global__ void rope_inplace(float* __restrict__ q, float* __restrict__ k,
                             const float* __restrict__ cs,
                             const float* __restrict__ sn) {
  const size_t total = (size_t)NB * NH * NT * (NHD / 2);
  size_t i = (size_t)blockIdx.x * blockDim.x + threadIdx.x;
  if (i >= total) return;
  const int d       = (int)(i & (NHD / 2 - 1));
  const size_t rest = i >> 6;
  const int t       = (int)(rest & (NT - 1));
  const size_t bh   = rest >> 11;
  const size_t base = (bh * NT + t) * NHD;
  const float c1 = cs[t * NHD + d],      c2 = cs[t * NHD + d + 64];
  const float s1 = sn[t * NHD + d],      s2 = sn[t * NHD + d + 64];
  const float q1 = q[base + d], q2 = q[base + d + 64];
  q[base + d]      = q1 * c1 - q2 * s1;
  q[base + d + 64] = q2 * c2 + q1 * s2;
  const float k1 = k[base + d], k2 = k[base + d + 64];
  k[base + d]      = k1 * c1 - k2 * s1;
  k[base + d + 64] = k2 * c2 + k1 * s2;
}

// ---------------------------------------------------------------------------
// Flash attention.  Dynamic LDS (bytes):
//   [0      ) Kpack  4ks x 8nt x 32 x 16 bf16                 32768
//   [32768  ) Vpack  4ks x 8nt x 32 x 16 bf16                 32768
//   [65536  ) Sf     128 x 132 f32  (aliases Qpack staging)   67584
//   [133120 ) Ppack  8mt x 4ks x 32 x 16 bf16                 32768
//   [165888 ) rowM/rowL/rowA 3 x 128 f32                       1536
// total 167424 B (< 320 KB/WGP -> one fat block per WGP)
// ---------------------------------------------------------------------------
__global__ __launch_bounds__(256) void attn_fa(
    const float* __restrict__ Qp, const float* __restrict__ Kp,
    const float* __restrict__ Vp, float* __restrict__ Op) {
  extern __shared__ char smem[];
  __bf16* Kpack = reinterpret_cast<__bf16*>(smem);
  __bf16* Vpack = reinterpret_cast<__bf16*>(smem + 32768);
  float*  Sf    = reinterpret_cast<float*>(smem + 65536);
  __bf16* Qpack = reinterpret_cast<__bf16*>(smem + 65536);   // alias
  __bf16* Ppack = reinterpret_cast<__bf16*>(smem + 133120);
  float* rowM = reinterpret_cast<float*>(smem + 165888);
  float* rowL = rowM + 128;
  float* rowA = rowL + 128;

  const int tid  = threadIdx.x;
  const int lane = tid & 31;
  const int half = lane >> 4;
  const int idx  = lane & 15;
  const int wave = tid >> 5;
  const int wm   = wave >> 1;
  const int wn   = wave & 1;

  const int bh   = blockIdx.y;
  const int qBlk = blockIdx.x * 128;
  const size_t headOff = (size_t)bh * NT * NHD;

  {  // stage Q into packed A-fragment layout [mtile][ks][lane][16]
    const int row = tid >> 1;
    const int seg = (tid & 1) * 64;
    const float* qp = Qp + headOff + (size_t)(qBlk + row) * NHD + seg;
#pragma unroll
    for (int c = 0; c < 8; ++c) {
      const int hd0 = seg + c * 8;
      float4 v0 = *reinterpret_cast<const float4*>(qp + c * 8);
      float4 v1 = *reinterpret_cast<const float4*>(qp + c * 8 + 4);
      v8bf pv;
      pv[0] = (__bf16)v0.x; pv[1] = (__bf16)v0.y; pv[2] = (__bf16)v0.z; pv[3] = (__bf16)v0.w;
      pv[4] = (__bf16)v1.x; pv[5] = (__bf16)v1.y; pv[6] = (__bf16)v1.z; pv[7] = (__bf16)v1.w;
      const int ks = hd0 >> 5;
      const int kk = hd0 & 31;
      const int hf = kk >> 4;
      const int j0 = kk & 15;
      *reinterpret_cast<v8bf*>(
          Qpack + (((row >> 4) * 4 + ks) * 32 + 16 * hf + (row & 15)) * 16 + j0) = pv;
    }
  }
  if (tid < 128) { rowM[tid] = -3.0e38f; rowL[tid] = 0.0f; }
  __syncthreads();

  // Q fragments live in registers for the whole kernel
  v16bf qf[2][4];
#pragma unroll
  for (int i = 0; i < 2; ++i)
#pragma unroll
    for (int ks = 0; ks < 4; ++ks)
      qf[i][ks] = *reinterpret_cast<const v16bf*>(
          Qpack + (((wm * 2 + i) * 4 + ks) * 32 + lane) * 16);

  const v8f zero = {};
  v8f oacc[2][4];
#pragma unroll
  for (int i = 0; i < 2; ++i)
#pragma unroll
    for (int n = 0; n < 4; ++n) oacc[i][n] = zero;

  const float scale = 0.08838834764831845f;  // HD^-0.5

  for (int jt = 0; jt <= (int)blockIdx.x; ++jt) {
    const int kvBlk = jt * 128;
    {  // stage K: row loads, packed B-layout stores  [ks][kv-tile][lane][16]
      const int kv  = tid >> 1;
      const int seg = (tid & 1) * 64;
      const float* kp = Kp + headOff + (size_t)(kvBlk + kv) * NHD + seg;
#pragma unroll
      for (int c = 0; c < 4; ++c) {
        const int hd0 = seg + c * 16;
        float4 v0 = *reinterpret_cast<const float4*>(kp + c * 16);
        float4 v1 = *reinterpret_cast<const float4*>(kp + c * 16 + 4);
        float4 v2 = *reinterpret_cast<const float4*>(kp + c * 16 + 8);
        float4 v3 = *reinterpret_cast<const float4*>(kp + c * 16 + 12);
        v8bf p0, p1;
        p0[0] = (__bf16)v0.x; p0[1] = (__bf16)v0.y; p0[2] = (__bf16)v0.z; p0[3] = (__bf16)v0.w;
        p0[4] = (__bf16)v1.x; p0[5] = (__bf16)v1.y; p0[6] = (__bf16)v1.z; p0[7] = (__bf16)v1.w;
        p1[0] = (__bf16)v2.x; p1[1] = (__bf16)v2.y; p1[2] = (__bf16)v2.z; p1[3] = (__bf16)v2.w;
        p1[4] = (__bf16)v3.x; p1[5] = (__bf16)v3.y; p1[6] = (__bf16)v3.z; p1[7] = (__bf16)v3.w;
        const int ks = hd0 >> 5;
        const int hf = (hd0 >> 4) & 1;
        __bf16* dst =
            Kpack + ((ks * 8 + (kv >> 4)) * 32 + 16 * hf + (kv & 15)) * 16;
        *reinterpret_cast<v8bf*>(dst)     = p0;
        *reinterpret_cast<v8bf*>(dst + 8) = p1;
      }
    }
    {  // stage V: column loads, packed B-layout stores [ks][hd-tile][lane][16]
      const int vhd = tid & 127;
      const int vkh = tid >> 7;
#pragma unroll
      for (int c = 0; c < 4; ++c) {
        const int kv0 = vkh * 64 + c * 16;
        const float* vp = Vp + headOff + (size_t)(kvBlk + kv0) * NHD + vhd;
        float cv[16];
#pragma unroll
        for (int u = 0; u < 16; ++u) cv[u] = vp[(size_t)u * NHD];
        v8bf p0, p1;
#pragma unroll
        for (int u = 0; u < 8; ++u) { p0[u] = (__bf16)cv[u]; p1[u] = (__bf16)cv[u + 8]; }
        const int ks = kv0 >> 5;
        const int hf = (kv0 >> 4) & 1;
        __bf16* dst =
            Vpack + ((ks * 8 + (vhd >> 4)) * 32 + 16 * hf + (vhd & 15)) * 16;
        *reinterpret_cast<v8bf*>(dst)     = p0;
        *reinterpret_cast<v8bf*>(dst + 8) = p1;
      }
    }
    __syncthreads();

    // S = Q * K^T
    v8f sacc[2][4];
#pragma unroll
    for (int i = 0; i < 2; ++i)
#pragma unroll
      for (int n = 0; n < 4; ++n) sacc[i][n] = zero;
#pragma unroll
    for (int ks = 0; ks < 4; ++ks) {
      v16bf kf[4];
#pragma unroll
      for (int n = 0; n < 4; ++n)
        kf[n] = *reinterpret_cast<const v16bf*>(
            Kpack + ((ks * 8 + wn * 4 + n) * 32 + lane) * 16);
#pragma unroll
      for (int i = 0; i < 2; ++i)
#pragma unroll
        for (int n = 0; n < 4; ++n)
          sacc[i][n] = __builtin_amdgcn_wmma_f32_16x16x32_bf16(
              false, qf[i][ks], false, kf[n], (short)0, sacc[i][n], false, false);
    }

    // scaled + causal-masked scores to LDS
#pragma unroll
    for (int i = 0; i < 2; ++i) {
      const int row0 = wm * 32 + i * 16 + 8 * half;
#pragma unroll
      for (int n = 0; n < 4; ++n) {
        const int col = wn * 64 + n * 16 + idx;
#pragma unroll
        for (int r = 0; r < 8; ++r) {
          const int row = row0 + r;
          float val = sacc[i][n][r] * scale;
          if (kvBlk + col > qBlk + row) val = -1.0e30f;
          Sf[row * 132 + col] = val;
        }
      }
    }
    __syncthreads();

    // online softmax: one thread per query row; P written in packed A-layout
    if (tid < 128) {
      const int r = tid;
      const float mOld = rowM[r];
      float mNew = mOld;
#pragma unroll
      for (int c = 0; c < 128; c += 4) {
        float4 v = *reinterpret_cast<const float4*>(&Sf[r * 132 + c]);
        mNew = fmaxf(mNew, fmaxf(fmaxf(v.x, v.y), fmaxf(v.z, v.w)));
      }
      float sum = 0.0f;
#pragma unroll
      for (int c8 = 0; c8 < 16; ++c8) {
        const int c0 = c8 * 8;
        float4 v0 = *reinterpret_cast<const float4*>(&Sf[r * 132 + c0]);
        float4 v1 = *reinterpret_cast<const float4*>(&Sf[r * 132 + c0 + 4]);
        float p[8];
        p[0] = __expf(v0.x - mNew); p[1] = __expf(v0.y - mNew);
        p[2] = __expf(v0.z - mNew); p[3] = __expf(v0.w - mNew);
        p[4] = __expf(v1.x - mNew); p[5] = __expf(v1.y - mNew);
        p[6] = __expf(v1.z - mNew); p[7] = __expf(v1.w - mNew);
        v8bf pv;
#pragma unroll
        for (int u = 0; u < 8; ++u) { sum += p[u]; pv[u] = (__bf16)p[u]; }
        const int ks = c0 >> 5;
        const int kk = c0 & 31;
        const int hf = kk >> 4;
        const int j0 = kk & 15;
        *reinterpret_cast<v8bf*>(
            Ppack + (((r >> 4) * 4 + ks) * 32 + 16 * hf + (r & 15)) * 16 + j0) = pv;
      }
      const float alpha = __expf(mOld - mNew);
      rowA[r] = alpha;
      rowL[r] = rowL[r] * alpha + sum;
      rowM[r] = mNew;
    }
    __syncthreads();

    // rescale running O by alpha(row): vector LDS reads
#pragma unroll
    for (int i = 0; i < 2; ++i) {
      const v8f a = *reinterpret_cast<const v8f*>(rowA + wm * 32 + i * 16 + 8 * half);
#pragma unroll
      for (int n = 0; n < 4; ++n)
#pragma unroll
        for (int r = 0; r < 8; ++r) oacc[i][n][r] *= a[r];
    }

    // O += P * V
#pragma unroll
    for (int ks = 0; ks < 4; ++ks) {
      v16bf pf[2];
#pragma unroll
      for (int i = 0; i < 2; ++i)
        pf[i] = *reinterpret_cast<const v16bf*>(
            Ppack + (((wm * 2 + i) * 4 + ks) * 32 + lane) * 16);
      v16bf vf[4];
#pragma unroll
      for (int n = 0; n < 4; ++n)
        vf[n] = *reinterpret_cast<const v16bf*>(
            Vpack + ((ks * 8 + wn * 4 + n) * 32 + lane) * 16);
#pragma unroll
      for (int i = 0; i < 2; ++i)
#pragma unroll
        for (int n = 0; n < 4; ++n)
          oacc[i][n] = __builtin_amdgcn_wmma_f32_16x16x32_bf16(
              false, pf[i], false, vf[n], (short)0, oacc[i][n], false, false);
    }
    __syncthreads();
  }

  // normalize and write O into [B, T, H, HD]
  const int b = bh >> 4;
  const int h = bh & (NH - 1);
#pragma unroll
  for (int i = 0; i < 2; ++i) {
    const int row0 = wm * 32 + i * 16 + 8 * half;
    const v8f l = *reinterpret_cast<const v8f*>(rowL + row0);
#pragma unroll
    for (int n = 0; n < 4; ++n) {
      const int hd = wn * 64 + n * 16 + idx;
#pragma unroll
      for (int r = 0; r < 8; ++r) {
        const int row = row0 + r;
        const float val = oacc[i][n][r] / l[r];
        const size_t t = (size_t)qBlk + row;
        Op[(((size_t)b * NT + t) * NH + h) * NHD + hd] = val;
      }
    }
  }
}

// ---------------------------------------------------------------------------
extern "C" void kernel_launch(void* const* d_in, const int* in_sizes, int n_in,
                              void* d_out, int out_size, void* d_ws,
                              size_t ws_size, hipStream_t stream) {
  const float* x  = (const float*)d_in[0];
  const float* cs = (const float*)d_in[1];
  const float* sn = (const float*)d_in[2];
  const float* Wq = (const float*)d_in[3];
  const float* Wk = (const float*)d_in[4];
  const float* Wv = (const float*)d_in[5];
  const float* Wo = (const float*)d_in[6];

  float* out = (float*)d_out;                       // [B,T,DM]
  const size_t headTotal = (size_t)NB * NH * NT * NHD;
  float* kOut = out + (size_t)NB * NT * NDM;        // [B,H,T,HD]
  float* vOut = kOut + headTotal;                   // [B,H,T,HD]

  float* qWs    = (float*)d_ws;                     // [B,H,T,HD]
  float* attnWs = qWs + headTotal;                  // [B,T,H,HD]

  dim3 blk(256);
  dim3 gG(NDM / 128, (NB * NT) / 128);              // (16, 32)

  gemm128_bf16<1><<<gG, blk, 0, stream>>>(x, Wq, qWs,  NB * NT, NDM, NDM);
  gemm128_bf16<1><<<gG, blk, 0, stream>>>(x, Wk, kOut, NB * NT, NDM, NDM);
  gemm128_bf16<1><<<gG, blk, 0, stream>>>(x, Wv, vOut, NB * NT, NDM, NDM);

  const size_t ropeN = (size_t)NB * NH * NT * (NHD / 2);
  rope_inplace<<<dim3((unsigned)((ropeN + 255) / 256)), blk, 0, stream>>>(
      qWs, kOut, cs, sn);

  const int attnSmem = 167424;
  (void)hipFuncSetAttribute((const void*)attn_fa,
                            hipFuncAttributeMaxDynamicSharedMemorySize,
                            attnSmem);
  attn_fa<<<dim3(NT / 128, NB * NH), blk, attnSmem, stream>>>(qWs, kOut, vOut,
                                                              attnWs);

  gemm128_bf16<0><<<gG, blk, 0, stream>>>(attnWs, Wo, out, NB * NT, NDM, NDM);
}
